// ISAB_49924699848999
// MI455X (gfx1250) — compile-verified, matrix operand order checked by reading, not run on hardware
//
#include <hip/hip_runtime.h>
#include <hip/hip_bf16.h>

typedef __attribute__((ext_vector_type(16))) _Float16 v16h;
typedef __attribute__((ext_vector_type(8)))  float    v8f;

#define EPS 1e-5f

__device__ __forceinline__ float wave_sum32(float x) {
  #pragma unroll
  for (int m = 16; m >= 1; m >>= 1) x += __shfl_xor(x, m, 32);
  return x;
}

// ---------------- Pass 1: per-(b,c) reductions over HW=1024 ----------------
// query[b,c] = <q[b,c,:], wq_lin> + bq ; keyv = <k,wk_lin>+bk ; kk = <k,w_sra>
__global__ __launch_bounds__(256)
void k_rowreduce(const float* __restrict__ q, const float* __restrict__ k,
                 const float* __restrict__ wq, const float* __restrict__ bq,
                 const float* __restrict__ wk, const float* __restrict__ bk,
                 const float* __restrict__ wsra,
                 float* __restrict__ query_ws, float* __restrict__ keyv_ws,
                 float* __restrict__ kk_ws) {
  __shared__ float r0[256], r1[256], r2[256];
  const int row = blockIdx.x;          // b*256 + c
  const int t = threadIdx.x;
  const float4 qa = ((const float4*)(q + (size_t)row * 1024))[t];
  const float4 ka = ((const float4*)(k + (size_t)row * 1024))[t];
  const float4 wa = ((const float4*)wq)[t];
  const float4 wb = ((const float4*)wk)[t];
  const float4 wc = ((const float4*)wsra)[t];
  r0[t] = qa.x*wa.x + qa.y*wa.y + qa.z*wa.z + qa.w*wa.w;
  r1[t] = ka.x*wb.x + ka.y*wb.y + ka.z*wb.z + ka.w*wb.w;
  r2[t] = ka.x*wc.x + ka.y*wc.y + ka.z*wc.z + ka.w*wc.w;
  __syncthreads();
  #pragma unroll
  for (int s = 128; s > 0; s >>= 1) {
    if (t < s) { r0[t] += r0[t+s]; r1[t] += r1[t+s]; r2[t] += r2[t+s]; }
    __syncthreads();
  }
  if (t == 0) {
    query_ws[row] = r0[0] + bq[0];
    keyv_ws[row]  = r1[0] + bk[0];
    kk_ws[row]    = r2[0];
  }
}

// ---------------- Pass 2: WMMA mid kernel (one workgroup, 32 waves) --------
// GEMM1: T(256x32) = w_kconv(256x256) x KK(256x32) + b_kconv*S_w   (K-dim = c)
// GEMM2: U(256x32) = w_qconv^T(256x256) x T(256x32)                (K-dim = o)
// then v[b] = <b_qconv, T[:,b]> and the channel-LN branch (wave w <-> batch w)
__global__ __launch_bounds__(1024)
void k_mid(const float* __restrict__ kk_ws, const float* __restrict__ query_ws,
           const float* __restrict__ keyv_ws,
           const float* __restrict__ w_qconv, const float* __restrict__ b_qconv,
           const float* __restrict__ w_kconv, const float* __restrict__ b_kconv,
           const float* __restrict__ w_ch, const float* __restrict__ b_ch,
           const float* __restrict__ g_cra, const float* __restrict__ beta_cra,
           const float* __restrict__ w_sra,
           float* __restrict__ u_ws, float* __restrict__ v_ws,
           float* __restrict__ chv_ws) {
  __shared__ float lds_T[256 * 32];
  __shared__ float red[1024];
  __shared__ float sw_sh;
  const int t    = threadIdx.x;
  const int lane = t & 31;
  const int w    = t >> 5;          // wave id 0..31
  // ---- S_w = sum(w_sra) over 1024 elems ----
  red[t] = w_sra[t];
  __syncthreads();
  #pragma unroll
  for (int s = 512; s > 0; s >>= 1) {
    if (t < s) red[t] += red[t+s];
    __syncthreads();
  }
  if (t == 0) sw_sh = red[0];
  __syncthreads();
  const float S_w = sw_sh;

  const int mt  = w >> 1;           // M tile 0..15
  const int nt  = w & 1;            // N tile 0..1
  const int col = lane & 15;        // row/col within tile from lane%16
  const int hi  = (lane >> 4) & 1;  // lane group

  // ---- GEMM1 ----
  v8f acc = {0.f,0.f,0.f,0.f,0.f,0.f,0.f,0.f};
  for (int kt = 0; kt < 8; ++kt) {
    v16h a, bf;
    #pragma unroll
    for (int h = 0; h < 16; ++h) {
      const int kk = kt*32 + ((h < 8) ? 0 : 8) + hi*8 + h;   // ISA f16 A/B K-map
      a[h]  = (_Float16)w_kconv[(mt*16 + col)*256 + kk];     // A[M=o, K=c]
      bf[h] = (_Float16)kk_ws[(nt*16 + col)*256 + kk];       // B[K=c, N=b]
    }
    acc = __builtin_amdgcn_wmma_f32_16x16x32_f16(false, a, false, bf,
                                                 (short)0, acc, false, false);
  }
  #pragma unroll
  for (int r = 0; r < 8; ++r) {
    const int o    = mt*16 + hi*8 + r;   // C/D layout: M = r + (lane>=16 ? 8:0)
    const int bcol = nt*16 + col;
    lds_T[o*32 + bcol] = acc[r] + b_kconv[o] * S_w;
  }
  __syncthreads();

  // ---- GEMM2 ----
  v8f acc2 = {0.f,0.f,0.f,0.f,0.f,0.f,0.f,0.f};
  for (int kt = 0; kt < 8; ++kt) {
    v16h a, bf;
    #pragma unroll
    for (int h = 0; h < 16; ++h) {
      const int kk = kt*32 + ((h < 8) ? 0 : 8) + hi*8 + h;
      a[h]  = (_Float16)w_qconv[kk*256 + mt*16 + col];       // A[M=c, K=o] = Wq[o,c]
      bf[h] = (_Float16)lds_T[kk*32 + nt*16 + col];          // B[K=o, N=b] = T[o,b]
    }
    acc2 = __builtin_amdgcn_wmma_f32_16x16x32_f16(false, a, false, bf,
                                                  (short)0, acc2, false, false);
  }
  #pragma unroll
  for (int r = 0; r < 8; ++r) {
    const int c    = mt*16 + hi*8 + r;
    const int bcol = nt*16 + col;
    u_ws[bcol*256 + c] = acc2[r];
  }
  __syncthreads();

  // ---- per-batch scalars + channel branch: wave w handles batch b = w ----
  {
    const int b = w;
    float vb = 0.f;
    #pragma unroll
    for (int j = 0; j < 8; ++j) {
      const int o = lane + 32*j;
      vb += lds_T[o*32 + b] * b_qconv[o];
    }
    vb = wave_sum32(vb);
    if (lane == 0) v_ws[b] = vb;

    float sb = 0.f;
    #pragma unroll
    for (int j = 0; j < 8; ++j) {
      const int c = lane + 32*j;
      sb += keyv_ws[b*256 + c] * w_ch[c];
    }
    sb = wave_sum32(sb);

    const float bch = b_ch[0];
    float yv[8], s1 = 0.f, s2 = 0.f;
    #pragma unroll
    for (int j = 0; j < 8; ++j) {
      const int c = lane + 32*j;
      const float y = query_ws[b*256 + c] * sb + bch;
      yv[j] = y; s1 += y; s2 += y*y;
    }
    s1 = wave_sum32(s1); s2 = wave_sum32(s2);
    const float m   = s1 * (1.f/256.f);
    const float var = s2 * (1.f/256.f) - m*m;
    const float inv = rsqrtf(var + EPS);
    #pragma unroll
    for (int j = 0; j < 8; ++j) {
      const int c = lane + 32*j;
      chv_ws[b*256 + c] = (yv[j] - m) * inv * g_cra[c] + beta_cra[c];
    }
  }
}

// ---------------- Pass 3: sp partials (streams q once) ---------------------
// block = (b, p-tile of 256, c-chunk of 64); no atomics, 4 partials per (b,p)
__global__ __launch_bounds__(256)
void k_sp_partial(const float* __restrict__ q, const float* __restrict__ u_ws,
                  const float* __restrict__ v_ws, const float* __restrict__ b_sra,
                  float* __restrict__ sp_part) {
  __shared__ float us[64];
  const int id = blockIdx.x;              // 32 * 4 * 4 = 512 blocks
  const int b  = id >> 4;
  const int pt = (id >> 2) & 3;
  const int cc = id & 3;
  const int t  = threadIdx.x;
  if (t < 64) us[t] = u_ws[b*256 + cc*64 + t];
  __syncthreads();
  const int p = pt*256 + t;
  const float* qb = q + ((size_t)(b*256 + cc*64)) * 1024 + p;
  float acc = 0.f;
  #pragma unroll 8
  for (int c = 0; c < 64; ++c) acc += qb[(size_t)c * 1024] * us[c];
  if (cc == 0) acc += v_ws[b] + b_sra[0];
  sp_part[(size_t)(cc*32 + b) * 1024 + p] = acc;
}

// ---------------- Pass 4: spatial layernorm per batch ----------------------
__global__ __launch_bounds__(256)
void k_sp_ln(const float* __restrict__ sp_part, const float* __restrict__ g_sra,
             const float* __restrict__ beta_sra, float* __restrict__ sp_n) {
  __shared__ float rs[256], rq[256];
  const int b = blockIdx.x;
  const int t = threadIdx.x;
  float vals[4], s1 = 0.f, s2 = 0.f;
  #pragma unroll
  for (int j = 0; j < 4; ++j) {
    const int p = t + j*256;
    float s = sp_part[(size_t)(0*32 + b)*1024 + p]
            + sp_part[(size_t)(1*32 + b)*1024 + p]
            + sp_part[(size_t)(2*32 + b)*1024 + p]
            + sp_part[(size_t)(3*32 + b)*1024 + p];
    vals[j] = s; s1 += s; s2 += s*s;
  }
  rs[t] = s1; rq[t] = s2;
  __syncthreads();
  #pragma unroll
  for (int s = 128; s > 0; s >>= 1) {
    if (t < s) { rs[t] += rs[t+s]; rq[t] += rq[t+s]; }
    __syncthreads();
  }
  const float m   = rs[0] * (1.f/1024.f);
  const float var = rq[0] * (1.f/1024.f) - m*m;
  const float inv = rsqrtf(var + EPS);
  #pragma unroll
  for (int j = 0; j < 4; ++j) {
    const int p = t + j*256;
    sp_n[(size_t)b*1024 + p] = (vals[j] - m) * inv * g_sra[p] + beta_sra[p];
  }
}

// ---------------- Pass 5: out[b,c,p] = sp_n[b,p] * chv[b,c] ----------------
__global__ __launch_bounds__(256)
void k_out(const float* __restrict__ sp_n, const float* __restrict__ chv,
           float* __restrict__ out) {
  const int base = (blockIdx.x * 256 + threadIdx.x) * 4;  // total 8,388,608 floats
  const int b = base >> 18;
  const int c = (base >> 10) & 255;
  const int p = base & 1023;
  const float cw = chv[b*256 + c];
  const float4 s = *(const float4*)(sp_n + b*1024 + p);
  float4 o; o.x = s.x*cw; o.y = s.y*cw; o.z = s.z*cw; o.w = s.w*cw;
  *(float4*)(out + base) = o;
}

extern "C" void kernel_launch(void* const* d_in, const int* in_sizes, int n_in,
                              void* d_out, int out_size, void* d_ws, size_t ws_size,
                              hipStream_t stream) {
  const float* q        = (const float*)d_in[0];
  const float* k        = (const float*)d_in[1];
  const float* wq_lin   = (const float*)d_in[2];
  const float* bq_lin   = (const float*)d_in[3];
  const float* wk_lin   = (const float*)d_in[4];
  const float* bk_lin   = (const float*)d_in[5];
  const float* w_ch     = (const float*)d_in[6];
  const float* b_ch     = (const float*)d_in[7];
  const float* g_cra    = (const float*)d_in[8];
  const float* beta_cra = (const float*)d_in[9];
  const float* w_qconv  = (const float*)d_in[10];
  const float* b_qconv  = (const float*)d_in[11];
  const float* w_kconv  = (const float*)d_in[12];
  const float* b_kconv  = (const float*)d_in[13];
  const float* w_sra    = (const float*)d_in[14];
  const float* b_sra    = (const float*)d_in[15];
  const float* g_sra    = (const float*)d_in[16];
  const float* beta_sra = (const float*)d_in[17];
  float* out = (float*)d_out;

  float* ws       = (float*)d_ws;
  float* query_ws = ws;            // 8192
  float* keyv_ws  = ws + 8192;     // 8192
  float* kk_ws    = ws + 16384;    // 8192
  float* u_ws     = ws + 24576;    // 8192
  float* v_ws     = ws + 32768;    // 32 (padded to 64)
  float* chv_ws   = ws + 32832;    // 8192
  float* sp_part  = ws + 41024;    // 4*32768
  float* sp_n     = ws + 172096;   // 32768  (total ~820 KB)

  k_rowreduce<<<8192, 256, 0, stream>>>(q, k, wq_lin, bq_lin, wk_lin, bk_lin,
                                        w_sra, query_ws, keyv_ws, kk_ws);
  k_mid<<<1, 1024, 0, stream>>>(kk_ws, query_ws, keyv_ws, w_qconv, b_qconv,
                                w_kconv, b_kconv, w_ch, b_ch, g_cra, beta_cra,
                                w_sra, u_ws, v_ws, chv_ws);
  k_sp_partial<<<512, 256, 0, stream>>>(q, u_ws, v_ws, b_sra, sp_part);
  k_sp_ln<<<32, 256, 0, stream>>>(sp_part, g_sra, beta_sra, sp_n);
  k_out<<<8192, 256, 0, stream>>>(sp_n, chv_ws, out);
}
